// RelExLSTM_4148938408302
// MI455X (gfx1250) — compile-verified, matrix operand order checked by reading, not run on hardware
//
#include <hip/hip_runtime.h>
#include <hip/hip_bf16.h>
#include <math.h>
#include <stdint.h>

#define VOCAB  32000
#define EMB    300
#define KPAD   320          // EMB padded to multiple of 32 (bf16 WMMA K)
#define UNITS  256
#define GATES  1024         // 4*UNITS
#define LABELS 19
#define MAXLEN 128
#define BATCH  512
#define ROWS   (MAXLEN*BATCH)   // 65536 time-major rows

#define LPAD   40           // LDS row pitch for a 32-elem K-tile row (bank-conflict-free)

typedef __attribute__((ext_vector_type(16))) __bf16 v16bf;
typedef __attribute__((ext_vector_type(8)))  float  v8f;

union Frag { uint4 u[2]; v16bf v; };

__device__ __forceinline__ float sigm(float x) { return 1.f / (1.f + __expf(-x)); }

// ---------------------------------------------------------------------------
// Pack helpers: f32 [K,N] row-major -> bf16 [N,Kpad] (transpose + zero-pad K)
// ---------------------------------------------------------------------------
__global__ void pack_transpose(const float* __restrict__ src, __bf16* __restrict__ dst,
                               int K, int N, int Kpad) {
    int idx = blockIdx.x * blockDim.x + threadIdx.x;
    if (idx >= N * Kpad) return;
    int n = idx / Kpad, k = idx - n * Kpad;
    float v = (k < K) ? src[(size_t)k * N + n] : 0.f;
    dst[idx] = (__bf16)v;
}

// Embedding gather -> time-major bf16 activations X[(t*B+b), KPAD]
__global__ void embed_pack(const int* __restrict__ tokens, const float* __restrict__ emb,
                           __bf16* __restrict__ X) {
    int idx = blockIdx.x * blockDim.x + threadIdx.x;
    if (idx >= ROWS * KPAD) return;
    int row = idx / KPAD, e = idx - row * KPAD;
    int t = row / BATCH, b = row - t * BATCH;
    int tok = tokens[b * MAXLEN + t];
    float v = (e < EMB) ? emb[(size_t)tok * EMB + e] : 0.f;
    X[idx] = (__bf16)v;
}

__global__ void init_state(__bf16* __restrict__ hbf, float* __restrict__ c) {
    int idx = blockIdx.x * blockDim.x + threadIdx.x;
    if (idx >= 2 * BATCH * UNITS) return;
    hbf[idx] = (__bf16)0.f;
    c[idx] = 0.f;
}

// ---------------------------------------------------------------------------
// Block GEMM engine: 256 threads = 8 waves (2 waveRows x 4 waveCols).
// Block tile 64x256, wave tile 32x64 (2x4 WMMA tiles, 8 wmma / k-step).
// A: row-major bf16 [M, lda]; Bt: N-major bf16 [N, ldb].
// K-tiles (32 wide) double-buffered through LDS with one barrier per k-step:
//   issue next tile's global loads -> 8 WMMAs on current LDS buffer ->
//   ds_store next buffer -> barrier.
// LDS rows padded to LPAD=40 bf16 (20 dwords): lanes reading 16B at row
// stride 80B hit 16 distinct 4-bank-aligned groups -> conflict-free.
// bf16 16x32 A-frag layout (ISA 7.12.2): lane<16 -> row=lane, K={0..7,16..23};
// lane>=16 -> row=lane-16, K={8..15,24..31}  => two contiguous 16B reads.
// ---------------------------------------------------------------------------
__device__ __forceinline__ void block_gemm_64x256(
    const __bf16* __restrict__ A, int lda, int rowBase,
    const __bf16* __restrict__ Bt, int ldb, int colBase,
    int K, float* __restrict__ C, int ldc)
{
    __shared__ __bf16 lA[2][64 * LPAD];    //  5.0 KB x2
    __shared__ __bf16 lB[2][256 * LPAD];   // 20.0 KB x2

    const int t = threadIdx.x;
    const int wave = t >> 5, lane = t & 31;
    const int waveRow = wave >> 2;            // 0..1
    const int waveCol = wave & 3;             // 0..3
    const int half = lane >> 4, l16 = lane & 15;
    const int kb = half * 8;                  // K sub-offset within 32-tile

    // staging: thread t handles (row = t/4, 16B chunk = t%4)
    const int sRow = t >> 2;
    const int sOff = (t & 3) * 8;             // element offset within 32-K tile

    const __bf16* gA = A + (size_t)(rowBase + sRow) * lda + sOff;
    const __bf16* gB0 = Bt + (size_t)(colBase + sRow) * ldb + sOff;

    v8f acc[2][4] = {};

    // ---- stage k-tile 0
    {
        uint4 ra = *(const uint4*)(gA);
        uint4 rb[4];
#pragma unroll
        for (int i = 0; i < 4; ++i)
            rb[i] = *(const uint4*)(gB0 + (size_t)(i * 64) * ldb);
        *(uint4*)&lA[0][sRow * LPAD + sOff] = ra;
#pragma unroll
        for (int i = 0; i < 4; ++i)
            *(uint4*)&lB[0][(i * 64 + sRow) * LPAD + sOff] = rb[i];
    }
    __syncthreads();

    const int KT = K / 32;
    for (int kt = 0; kt < KT; ++kt) {
        const int cur = kt & 1, nxt = cur ^ 1;
        const bool have = (kt + 1) < KT;

        // issue next tile's global loads (latency overlapped with WMMAs below)
        uint4 ra; uint4 rb[4];
        if (have) {
            const int k0 = (kt + 1) * 32;
            ra = *(const uint4*)(gA + k0);
#pragma unroll
            for (int i = 0; i < 4; ++i)
                rb[i] = *(const uint4*)(gB0 + (size_t)(i * 64) * ldb + k0);
            if (kt + 2 < KT) {   // pre-warm caches two tiles ahead
                __builtin_prefetch(gA + (kt + 2) * 32, 0, 0);
                __builtin_prefetch(gB0 + (kt + 2) * 32, 0, 0);
            }
        }

        // fragments from current LDS buffer + 8 WMMAs
        Frag bfr[4];
#pragma unroll
        for (int j = 0; j < 4; ++j) {
            const int br = waveCol * 64 + j * 16 + l16;
            bfr[j].u[0] = *(const uint4*)&lB[cur][br * LPAD + kb];
            bfr[j].u[1] = *(const uint4*)&lB[cur][br * LPAD + kb + 16];
        }
#pragma unroll
        for (int rt = 0; rt < 2; ++rt) {
            const int ar = waveRow * 32 + rt * 16 + l16;
            Frag a;
            a.u[0] = *(const uint4*)&lA[cur][ar * LPAD + kb];
            a.u[1] = *(const uint4*)&lA[cur][ar * LPAD + kb + 16];
#pragma unroll
            for (int j = 0; j < 4; ++j)
                acc[rt][j] = __builtin_amdgcn_wmma_f32_16x16x32_bf16(
                    false, a.v, false, bfr[j].v, (short)0, acc[rt][j], false, false);
        }

        // commit next tile into the other buffer
        if (have) {
            *(uint4*)&lA[nxt][sRow * LPAD + sOff] = ra;
#pragma unroll
            for (int i = 0; i < 4; ++i)
                *(uint4*)&lB[nxt][(i * 64 + sRow) * LPAD + sOff] = rb[i];
        }
        __syncthreads();
    }

    // C/D layout: VGPR r -> row = r + 8*half, col = lane&15
#pragma unroll
    for (int rt = 0; rt < 2; ++rt) {
#pragma unroll
        for (int j = 0; j < 4; ++j) {
            const int col = colBase + waveCol * 64 + j * 16 + l16;
#pragma unroll
            for (int r = 0; r < 8; ++r) {
                const int row = rowBase + waveRow * 32 + rt * 16 + r + half * 8;
                C[(size_t)row * ldc + col] = acc[rt][j][r];
            }
        }
    }
}

// Time-parallel input projection: ZX[dir] = X @ W[dir]  (65536 x 1024, K=320)
// grid (1024, 4, 2), block 256
__global__ void gemm_xw(const __bf16* __restrict__ X,
                        const __bf16* __restrict__ WtF, const __bf16* __restrict__ WtB,
                        float* __restrict__ ZXf, float* __restrict__ ZXb) {
    const int rowBase = blockIdx.x * 64;
    const int colBase = blockIdx.y * 256;
    const __bf16* Bt = blockIdx.z ? WtB : WtF;
    float* Z = blockIdx.z ? ZXb : ZXf;
    block_gemm_64x256(X, KPAD, rowBase, Bt, KPAD, colBase, KPAD, Z, GATES);
}

// Recurrent projection for one step, both directions: zbuf[dir] = h[dir] @ U[dir]
// grid (8, 4, 2), block 256
__global__ void gemm_rec(const __bf16* __restrict__ hbf,
                         const __bf16* __restrict__ UtF, const __bf16* __restrict__ UtB,
                         float* __restrict__ zbuf) {
    const int dir = blockIdx.z;
    const int rowBase = blockIdx.x * 64;
    const int colBase = blockIdx.y * 256;
    const __bf16* Ah = hbf + (size_t)dir * BATCH * UNITS;
    const __bf16* Bt = dir ? UtB : UtF;
    float* Z = zbuf + (size_t)dir * BATCH * GATES;
    block_gemm_64x256(Ah, UNITS, rowBase, Bt, UNITS, colBase, UNITS, Z, GATES);
}

// Elementwise LSTM cell update (f32 math), both directions.
// z = zbuf (h@U) + ZX[t or T-1-t] (x@W) + bias; Keras gate order i,f,g,o.
__global__ void lstm_gates(const float* __restrict__ zbuf,
                           const float* __restrict__ ZXf, const float* __restrict__ ZXb,
                           const float* __restrict__ bf_, const float* __restrict__ bb_,
                           __bf16* __restrict__ hbf, float* __restrict__ c, int t) {
    int idx = blockIdx.x * blockDim.x + threadIdx.x;
    if (idx >= 2 * BATCH * UNITS) return;
    const int dir = idx / (BATCH * UNITS);
    const int r   = idx - dir * (BATCH * UNITS);
    const int b   = r / UNITS;
    const int u   = r - b * UNITS;
    const float* ZX   = dir ? ZXb : ZXf;
    const float* bias = dir ? bb_ : bf_;
    const int tt = dir ? (MAXLEN - 1 - t) : t;
    const size_t zr = (size_t)dir * BATCH * GATES + (size_t)b * GATES;
    const size_t xr = ((size_t)tt * BATCH + b) * GATES;

    float zi = zbuf[zr + u]       + ZX[xr + u]       + bias[u];
    float zf = zbuf[zr + 256 + u] + ZX[xr + 256 + u] + bias[256 + u];
    float zg = zbuf[zr + 512 + u] + ZX[xr + 512 + u] + bias[512 + u];
    float zo = zbuf[zr + 768 + u] + ZX[xr + 768 + u] + bias[768 + u];

    float cn = sigm(zf) * c[idx] + sigm(zi) * tanhf(zg);
    float hn = sigm(zo) * tanhf(cn);
    c[idx] = cn;
    hbf[idx] = (__bf16)hn;          // bf16: next step's WMMA A operand
}

// Dense (512->19) + softmax. One wave per batch row; lane l<19 owns label l.
__global__ void dense_softmax(const __bf16* __restrict__ hbf, const float* __restrict__ Wd,
                              const float* __restrict__ bd, float* __restrict__ out) {
    const int b = blockIdx.x;
    const int lane = threadIdx.x;
    const __bf16* hf = hbf + (size_t)b * UNITS;
    const __bf16* hb = hbf + (size_t)BATCH * UNITS + (size_t)b * UNITS;

    float acc = (lane < LABELS) ? bd[lane] : 0.f;
    for (int j = 0; j < UNITS; ++j) {
        float h = (float)hf[j];
        if (lane < LABELS) acc += h * Wd[(size_t)j * LABELS + lane];
    }
    for (int j = 0; j < UNITS; ++j) {
        float h = (float)hb[j];
        if (lane < LABELS) acc += h * Wd[(size_t)(UNITS + j) * LABELS + lane];
    }
    float m = (lane < LABELS) ? acc : -INFINITY;
    for (int off = 16; off > 0; off >>= 1) m = fmaxf(m, __shfl_xor(m, off, 32));
    float e = (lane < LABELS) ? __expf(acc - m) : 0.f;
    float s = e;
    for (int off = 16; off > 0; off >>= 1) s += __shfl_xor(s, off, 32);
    if (lane < LABELS) out[(size_t)b * LABELS + lane] = e / s;
}

// ---------------------------------------------------------------------------
extern "C" void kernel_launch(void* const* d_in, const int* in_sizes, int n_in,
                              void* d_out, int out_size, void* d_ws, size_t ws_size,
                              hipStream_t stream) {
    const int*   tokens = (const int*)  d_in[0];
    const float* emb    = (const float*)d_in[1];
    const float* W_f    = (const float*)d_in[2];
    const float* U_f    = (const float*)d_in[3];
    const float* b_f    = (const float*)d_in[4];
    const float* W_b    = (const float*)d_in[5];
    const float* U_b    = (const float*)d_in[6];
    const float* b_b    = (const float*)d_in[7];
    const float* Wd     = (const float*)d_in[8];
    const float* bd     = (const float*)d_in[9];
    float* out = (float*)d_out;

    uint8_t* ws = (uint8_t*)d_ws;
    auto carve = [&](size_t bytes) -> void* {
        void* p = ws;
        ws += (bytes + 255) & ~size_t(255);
        return p;
    };
    __bf16* WtF  = (__bf16*)carve((size_t)GATES * KPAD * 2);   // W_f^T padded
    __bf16* WtB  = (__bf16*)carve((size_t)GATES * KPAD * 2);
    __bf16* UtF  = (__bf16*)carve((size_t)GATES * UNITS * 2);  // U_f^T
    __bf16* UtB  = (__bf16*)carve((size_t)GATES * UNITS * 2);
    __bf16* Xbf  = (__bf16*)carve((size_t)ROWS * KPAD * 2);    // 40 MB
    float*  ZXf  = (float*) carve((size_t)ROWS * GATES * 4);   // 256 MB
    float*  ZXb  = (float*) carve((size_t)ROWS * GATES * 4);   // 256 MB
    float*  zbuf = (float*) carve((size_t)2 * BATCH * GATES * 4);
    __bf16* hbf  = (__bf16*)carve((size_t)2 * BATCH * UNITS * 2);
    float*  cbuf = (float*) carve((size_t)2 * BATCH * UNITS * 4);

    // 1) pack weights (transpose to N-major bf16, pad K for W)
    pack_transpose<<<(GATES * KPAD + 255) / 256, 256, 0, stream>>>(W_f, WtF, EMB, GATES, KPAD);
    pack_transpose<<<(GATES * KPAD + 255) / 256, 256, 0, stream>>>(W_b, WtB, EMB, GATES, KPAD);
    pack_transpose<<<(GATES * UNITS + 255) / 256, 256, 0, stream>>>(U_f, UtF, UNITS, GATES, UNITS);
    pack_transpose<<<(GATES * UNITS + 255) / 256, 256, 0, stream>>>(U_b, UtB, UNITS, GATES, UNITS);

    // 2) embedding gather -> time-major bf16 activations
    embed_pack<<<(ROWS * KPAD + 255) / 256, 256, 0, stream>>>(tokens, emb, Xbf);

    // 3) time-parallel input projections for both directions (bulk of the FLOPs)
    gemm_xw<<<dim3(ROWS / 64, GATES / 256, 2), 256, 0, stream>>>(Xbf, WtF, WtB, ZXf, ZXb);

    // 4) recurrence, fwd+bwd fused per step
    init_state<<<(2 * BATCH * UNITS + 255) / 256, 256, 0, stream>>>(hbf, cbuf);
    for (int t = 0; t < MAXLEN; ++t) {
        gemm_rec<<<dim3(BATCH / 64, GATES / 256, 2), 256, 0, stream>>>(hbf, UtF, UtB, zbuf);
        lstm_gates<<<(2 * BATCH * UNITS + 255) / 256, 256, 0, stream>>>(
            zbuf, ZXf, ZXb, b_f, b_b, hbf, cbuf, t);
    }

    // 5) dense + softmax
    dense_softmax<<<BATCH, 32, 0, stream>>>(hbf, Wd, bd, out);
}
